// CrossViewSwapAttention_8684423872895
// MI455X (gfx1250) — matrix operand order repaired and checked by reading, not compiled, not run on hardware
//
#include <hip/hip_runtime.h>
#include <hip/hip_bf16.h>

typedef _Float16 h16;
typedef __attribute__((ext_vector_type(16))) _Float16 v16h;
typedef __attribute__((ext_vector_type(8)))  _Float16 v8h;
typedef __attribute__((ext_vector_type(8)))  float    v8f;

#define DEV __device__ __forceinline__

// ---------------- WMMA fragment helpers (gfx1250, wave32) ----------------
// A (16x32 f16, row-major source M x K):
//   lanes 0-15 : row = lane,    halves 0-15 = K[k0 .. k0+15]
//   lanes 16-31: row = lane-16, halves 0-15 = K[k0+16 .. k0+31]
DEV v16h load_a_frag(const h16* base, int stride, int k0, int lane) {
  const h16* p = base + (size_t)(lane & 15) * stride + k0 + ((lane & 16) ? 16 : 0);
  v8h lo = *(const v8h*)(p);
  v8h hi = *(const v8h*)(p + 8);
  v16h r;
#pragma unroll
  for (int i = 0; i < 8; ++i) { r[i] = lo[i]; r[i + 8] = hi[i]; }
  return r;
}
// B (32x16 f16, source stored N x K row-major, i.e. weight[out][in]):
//   lanes 0-15 : n = lane,    halves 0-15 = K[k0 .. k0+15]
//   lanes 16-31: n = lane-16, halves 0-15 = K[k0+16 .. k0+31]
DEV v16h load_b_frag(const h16* base, int stride, int k0, int lane) {
  const h16* p = base + (size_t)(lane & 15) * stride + k0 + ((lane & 16) ? 16 : 0);
  v8h lo = *(const v8h*)(p);
  v8h hi = *(const v8h*)(p + 8);
  v16h r;
#pragma unroll
  for (int i = 0; i < 8; ++i) { r[i] = lo[i]; r[i + 8] = hi[i]; }
  return r;
}
DEV v8f wmma16(v16h a, v16h b, v8f c) {
  return __builtin_amdgcn_wmma_f32_16x16x32_f16(false, a, false, b, (short)0, c, false, false);
}
DEV float block_sum128(float v, float* sm) {
  int t = threadIdx.x;
  sm[t] = v; __syncthreads();
  for (int s = 64; s > 0; s >>= 1) { if (t < s) sm[t] += sm[t + s]; __syncthreads(); }
  float r = sm[0]; __syncthreads();
  return r;
}
DEV float gelu_exact(float v) { return 0.5f * v * (1.0f + erff(v * 0.70710678118f)); }

// ---------------- small elementwise kernels ----------------
__global__ void k_cvt(const float* s, h16* d, int n) {
  int i = blockIdx.x * 256 + threadIdx.x;
  if (i < n) d[i] = (h16)s[i];
}

// gather x (1,128,128,128) channel-major -> window-token-major xw[tok][d]
__global__ void k_gather_x(const float* x, float* xw) {
  int i = blockIdx.x * 256 + threadIdx.x;            // tok*128 + d
  int tok = i >> 7, d = i & 127;
  int l = tok >> 8, qt = tok & 255;
  int r = (l >> 3) * 16 + (qt >> 4), c = (l & 7) * 16 + (qt & 15);
  xw[i] = x[d * 16384 + r * 128 + c];
}

// image-plane embedding, normalized per pixel over 128 dims.
__global__ void k_embed(const float* I_inv, const float* E_inv,
                        const float* img_W, const float* cam_W, float* img_embed) {
  __shared__ float sm[128];
  int blk = blockIdx.x;                 // cam*1024 + p
  int cam = blk >> 10, p = blk & 1023;
  int d = threadIdx.x;
  int i = p >> 5, j = p & 31;
  float fx = j * (256.0f / 31.0f), fy = i * (256.0f / 31.0f);
  const float* I = I_inv + cam * 9;
  const float* E = E_inv + cam * 16;
  float c0 = I[0] * fx + I[1] * fy + I[2];
  float c1 = I[3] * fx + I[4] * fy + I[5];
  float c2 = I[6] * fx + I[7] * fy + I[8];
  float dv[4];
#pragma unroll
  for (int rr = 0; rr < 4; ++rr)
    dv[rr] = E[rr*4+0]*c0 + E[rr*4+1]*c1 + E[rr*4+2]*c2 + E[rr*4+3];
  float ce = 0.f, de = 0.f;
#pragma unroll
  for (int cc = 0; cc < 4; ++cc) {
    ce += cam_W[d*4+cc] * E[cc*4+3];
    de += img_W[d*4+cc] * dv[cc];
  }
  float e = de - ce;
  float ss = block_sum128(e*e, sm);
  img_embed[(size_t)blk*128 + d] = e / (sqrtf(ss) + 1e-7f);
}

// BN(eval)+ReLU for both fp and fl branches, channel-major feature -> token-major f16
__global__ void k_featnorm(const float* feat,
    const float* gfp, const float* bfp, const float* mfp, const float* vfp,
    const float* gfl, const float* bfl, const float* mfl, const float* vfl,
    h16* hfp, h16* hfl) {
  int i = blockIdx.x * 256 + threadIdx.x;            // 6144*128
  int ct = i >> 7, c = i & 127;
  int cam = ct >> 10, p = ct & 1023;
  float f = feat[((size_t)cam*128 + c)*1024 + p];
  float a = (f - mfp[c]) * rsqrtf(vfp[c] + 1e-5f) * gfp[c] + bfp[c];
  hfp[i] = (h16)fmaxf(a, 0.f);
  float b = (f - mfl[c]) * rsqrtf(vfl[c] + 1e-5f) * gfl[c] + bfl[c];
  hfl[i] = (h16)fmaxf(b, 0.f);
}

// query token: bev embed (normalized) + x, then LN(qg,qb) -> f16 tokens
__global__ void k_qtok(const float* grid0, const float* xw,
                       const float* E_inv, const float* cam_W,
                       const float* bev_W, const float* bev_b,
                       const float* qg, const float* qb, h16* qln) {
  __shared__ float sm[128];
  int blk = blockIdx.x;                 // cam*16384 + tok
  int cam = blk >> 14, tok = blk & 16383;
  int l = tok >> 8, qt = tok & 255;
  int r = (l >> 3) * 16 + (qt >> 4), c = (l & 7) * 16 + (qt & 15);
  int d = threadIdx.x;
  const float* E = E_inv + cam * 16;
  float ce = 0.f;
#pragma unroll
  for (int cc = 0; cc < 4; ++cc) ce += cam_W[d*4+cc] * E[cc*4+3];
  float g0 = grid0[r*128 + c], g1 = grid0[16384 + r*128 + c];
  float w = bev_W[d*2]*g0 + bev_W[d*2+1]*g1 + bev_b[d] - ce;
  float ss = block_sum128(w*w, sm);
  float q = w / (sqrtf(ss) + 1e-7f) + xw[(size_t)tok*128 + d];
  float mu = block_sum128(q, sm) * (1.f/128.f);
  float t = q - mu;
  float vv = block_sum128(t*t, sm) * (1.f/128.f);
  qln[(size_t)blk*128 + d] = (h16)(t * rsqrtf(vv + 1e-5f) * qg[d] + qb[d]);
}

// generic LayerNorm (f32 tokens -> f16 tokens), DIM=128
__global__ void k_ln(const float* in, const float* g, const float* b, h16* out) {
  __shared__ float sm[128];
  int tok = blockIdx.x, d = threadIdx.x;
  float v = in[(size_t)tok*128 + d];
  float mu = block_sum128(v, sm) * (1.f/128.f);
  float t = v - mu;
  float vv = block_sum128(t*t, sm) * (1.f/128.f);
  out[(size_t)tok*128 + d] = (h16)(t * rsqrtf(vv + 1e-5f) * g[d] + b[d]);
}

// final LN + scatter window-token order back to (b,DIM,H,W)
__global__ void k_final(const float* in, const float* g, const float* b, float* out) {
  __shared__ float sm[128];
  int tok = blockIdx.x, d = threadIdx.x;
  float v = in[(size_t)tok*128 + d];
  float mu = block_sum128(v, sm) * (1.f/128.f);
  float t = v - mu;
  float vv = block_sum128(t*t, sm) * (1.f/128.f);
  float y = t * rsqrtf(vv + 1e-5f) * g[d] + b[d];
  int l = tok >> 8, qt = tok & 255;
  int r = (l >> 3) * 16 + (qt >> 4), c = (l & 7) * 16 + (qt & 15);
  out[d * 16384 + r * 128 + c] = y;
}

// ---------------- generic WMMA GEMM: out = A(MxK,f16) * B(NxK,f16)^T ----------------
// modes: 0 = f32 out (+bias,+res)   1 = f16 out (+bias)
//        2 = gelu(f16) out (+bias)  3 = scatter Q-heads  4 = scatter K-heads  5 = scatter V^T
__global__ void k_gemm(const h16* __restrict__ A, const h16* __restrict__ B,
                       const float* bias, const float* res,
                       float* outF, h16* outH,
                       int M, int N, int K, int mode, int stage, int cams) {
  int lane = threadIdx.x & 31, wv = threadIdx.x >> 5;
  int M0 = blockIdx.x * 64 + wv * 16;
  int N0 = blockIdx.y * 16;
  const h16* Ab = A + (size_t)M0 * K;
  const h16* Bb = B + (size_t)N0 * K;
  v8f acc = {};
  for (int k0 = 0; k0 < K; k0 += 32) {
    v16h a = load_a_frag(Ab, K, k0, lane);
    v16h b = load_b_frag(Bb, K, k0, lane);
    acc = wmma16(a, b, acc);
  }
  int n = N0 + (lane & 15);
  int mb = (lane & 16) ? 8 : 0;
#pragma unroll
  for (int rI = 0; rI < 8; ++rI) {
    int m = M0 + mb + rI;
    float v = acc[rI];
    if (bias) v += bias[n];
    if (mode == 0) {
      if (res) v += res[(size_t)m * N + n];
      outF[(size_t)m * N + n] = v;
    } else if (mode == 1) {
      outH[(size_t)m * N + n] = (h16)v;
    } else if (mode == 2) {
      outH[(size_t)m * N + n] = (h16)gelu_exact(v);
    } else if (mode == 3) {           // Q: rows=(cam,tok) cols=(head,dh) -> [l][cam][h][256][32]
      int cam = m >> 14, tok = m & 16383;
      int l = tok >> 8, qt = tok & 255, hm = n >> 5, dh = n & 31;
      size_t idx = ((((size_t)l * cams + cam) * 4 + hm) * 256 + qt) * 32 + dh;
      outH[idx] = (h16)v;
    } else {                          // K / V^T: rows=(cam,pos) -> window layout
      int cam = m >> 10, p = m & 1023, y = p >> 5, xx = p & 31, l, tt;
      if (stage == 1) { l = (y >> 2) * 8 + (xx >> 2); tt = cam * 16 + (y & 3) * 4 + (xx & 3); }
      else           { l = (y & 7) * 8 + (xx & 7);   tt = cam * 16 + (y >> 3) * 4 + (xx >> 3); }
      int hm = n >> 5, dh = n & 31;
      size_t idx = (mode == 4) ? (((size_t)l * 4 + hm) * 96 + tt) * 32 + dh
                               : (((size_t)l * 4 + hm) * 32 + dh) * 96 + tt;
      outH[idx] = (h16)v;
    }
  }
}

// ---------------- fused windowed attention (per window, per camera) ----------------
// QH: [l][cams][4][256][32]  KH: [l][4][96][32]  VT: [l][4][32][96]
// Aout (f16): [l][cams][256][128]  (heads concatenated)
__global__ void k_attn(const h16* __restrict__ QH, const h16* __restrict__ KH,
                       const h16* __restrict__ VT, const float* logit,
                       h16* Aout, int cams) {
  __shared__ float sc[8][1536];                           // 16 x 96 scores per wave
  __shared__ __attribute__((aligned(16))) h16 pb[8][1536]; // 16 x 96 probs per wave
  int l = blockIdx.x, cam = blockIdx.y;
  int lane = threadIdx.x & 31, wv = threadIdx.x >> 5;
  int mb = (lane & 16) ? 8 : 0, nn = lane & 15;
  for (int task = wv; task < 64; task += 8) {
    int hm = task >> 4, q16 = task & 15;
    const h16* Qb = QH + ((((size_t)l * cams + cam) * 4 + hm) * 256 + q16 * 16) * 32;
    v16h aq = load_a_frag(Qb, 32, 0, lane);
    float scale = 0.176776695f * __expf(logit[hm]);       // dh^-0.5 * exp(logit)
    // scores: 6 WMMAs of 16x16, K = 32 (one step)
    for (int nt = 0; nt < 6; ++nt) {
      v16h bk = load_b_frag(KH + (((size_t)l * 4 + hm) * 96 + nt * 16) * 32, 32, 0, lane);
      v8f c = {};
      c = wmma16(aq, bk, c);
#pragma unroll
      for (int rI = 0; rI < 8; ++rI)
        sc[wv][(mb + rI) * 96 + nt * 16 + nn] = c[rI] * scale;
    }
    // softmax over 96 keys; lanes 0-15 each own one row
    if (lane < 16) {
      float* rp = &sc[wv][lane * 96];
      h16*  pp = &pb[wv][lane * 96];
      float mx = -1e30f;
      for (int k = 0; k < 96; ++k) mx = fmaxf(mx, rp[k]);
      float s = 0.f;
      for (int k = 0; k < 96; ++k) { float e = __expf(rp[k] - mx); s += e; pp[k] = (h16)e; }
      float inv = 1.f / s;
      for (int k = 0; k < 96; ++k) pp[k] = (h16)((float)pp[k] * inv);
    }
    // PV: out 16x32 = 2 N-tiles, K = 96 -> 3 WMMA steps each
    for (int nt2 = 0; nt2 < 2; ++nt2) {
      v8f acc = {};
      for (int ks = 0; ks < 3; ++ks) {
        v16h ap = load_a_frag(&pb[wv][0], 96, ks * 32, lane);
        v16h bv = load_b_frag(VT + (((size_t)l * 4 + hm) * 32 + nt2 * 16) * 96, 96, ks * 32, lane);
        acc = wmma16(ap, bv, acc);
      }
#pragma unroll
      for (int rI = 0; rI < 8; ++rI) {
        int q = q16 * 16 + mb + rI;
        int col = hm * 32 + nt2 * 16 + nn;
        Aout[(((size_t)l * cams + cam) * 256 + q) * 128 + col] = (h16)acc[rI];
      }
    }
  }
}

// per-window camera weighting: mean over 256 positions, LN, MLP(128->64->1), softmax over 6 cams
__global__ void k_camweight(const h16* A1, const float* cs_g, const float* cs_b,
                            const float* W1, const float* b1,
                            const float* W2, const float* b2, float* cw) {
  __shared__ float md[6][128];
  __shared__ float lg[6];
  int l = blockIdx.x, d = threadIdx.x;
  for (int cam = 0; cam < 6; ++cam) {
    float s = 0.f;
    const h16* p = A1 + (((size_t)l * 6 + cam) * 256) * 128 + d;
    for (int qt = 0; qt < 256; ++qt) s += (float)p[qt * 128];
    md[cam][d] = s * (1.f / 256.f);
  }
  __syncthreads();
  if (d < 6) {
    int cam = d;
    float mu = 0.f;
    for (int k = 0; k < 128; ++k) mu += md[cam][k];
    mu *= (1.f / 128.f);
    float vv = 0.f;
    for (int k = 0; k < 128; ++k) { float t = md[cam][k] - mu; vv += t * t; }
    vv *= (1.f / 128.f);
    float rs = rsqrtf(vv + 1e-5f);
    float logit = b2[0];
    for (int j = 0; j < 64; ++j) {
      float h = b1[j];
      for (int k = 0; k < 128; ++k) {
        float xn = (md[cam][k] - mu) * rs * cs_g[k] + cs_b[k];
        h += W1[j * 128 + k] * xn;
      }
      logit += W2[j] * gelu_exact(h);
    }
    lg[cam] = logit;
  }
  __syncthreads();
  if (d == 0) {
    float mx = -1e30f;
    for (int cam = 0; cam < 6; ++cam) mx = fmaxf(mx, lg[cam]);
    float e[6], s = 0.f;
    for (int cam = 0; cam < 6; ++cam) { e[cam] = __expf(lg[cam] - mx); s += e[cam]; }
    for (int cam = 0; cam < 6; ++cam) cw[l * 6 + cam] = e[cam] / s;
  }
}

__global__ void k_camsum(const h16* A1, const float* cw, h16* awh) {
  int i = blockIdx.x * 256 + threadIdx.x;     // 16384*128
  int tok = i >> 7, d = i & 127, l = tok >> 8, qt = tok & 255;
  float s = 0.f;
  for (int cam = 0; cam < 6; ++cam)
    s += cw[l * 6 + cam] * (float)A1[(((size_t)l * 6 + cam) * 256 + qt) * 128 + d];
  awh[i] = (h16)s;
}

// ---------------- host ----------------
extern "C" void kernel_launch(void* const* d_in, const int* in_sizes, int n_in,
                              void* d_out, int out_size, void* d_ws, size_t ws_size,
                              hipStream_t stream) {
  (void)in_sizes; (void)n_in; (void)out_size; (void)ws_size;
  const float* x       = (const float*)d_in[0];
  const float* feature = (const float*)d_in[1];
  const float* I_inv   = (const float*)d_in[2];
  const float* E_inv   = (const float*)d_in[3];
  const float* grid0   = (const float*)d_in[4];
  auto P = [&](int i) { return (const float*)d_in[i]; };

  char* base = (char*)d_ws;
  size_t off = 0;
  auto alloc = [&](size_t bytes) -> void* {
    void* r = base + off;
    off = (off + bytes + 255) & ~(size_t)255;
    return r;
  };
  auto cvt = [&](const float* s, h16* d, int n) {
    k_cvt<<<(n + 255) / 256, 256, 0, stream>>>(s, d, n);
  };

  // persistent buffers
  float* xw   = (float*)alloc((size_t)16384 * 128 * 4);
  float* imge = (float*)alloc((size_t)6144 * 128 * 4);
  h16*   hfp  = (h16*)  alloc((size_t)6144 * 128 * 2);
  h16*   hfl  = (h16*)  alloc((size_t)6144 * 128 * 2);
  float* ktok = (float*)alloc((size_t)6144 * 128 * 4);
  float* vtok = (float*)alloc((size_t)6144 * 128 * 4);
  float* out1 = (float*)alloc((size_t)16384 * 128 * 4);
  float* xs1  = (float*)alloc((size_t)16384 * 128 * 4);
  float* o2a  = (float*)alloc((size_t)16384 * 128 * 4);
  float* o2b  = (float*)alloc((size_t)16384 * 128 * 4);
  float* cw   = (float*)alloc(64 * 6 * 4);

  // f16 weights: fp.W, fl.W, Wq1,Wk1,Wv1,Wp1, mlp1.W1, mlp1.W2, Wq2,Wk2,Wv2,Wp2, mlp2.W1, mlp2.W2
  const int widx[14] = {14, 9, 21, 25, 29, 31, 67, 69, 42, 46, 50, 52, 71, 73};
  const int wn[14]   = {16384,16384,16384,16384,16384,16384,32768,32768,
                        16384,16384,16384,16384,32768,32768};
  h16* Wf[14];
  for (int i = 0; i < 14; ++i) { Wf[i] = (h16*)alloc((size_t)wn[i] * 2); cvt(P(widx[i]), Wf[i], wn[i]); }
  h16 *FPW = Wf[0], *FLW = Wf[1], *Wq1 = Wf[2], *Wk1 = Wf[3], *Wv1 = Wf[4], *Wp1 = Wf[5];
  h16 *M1W1 = Wf[6], *M1W2 = Wf[7], *Wq2 = Wf[8], *Wk2 = Wf[9], *Wv2 = Wf[10], *Wp2 = Wf[11];
  h16 *M2W1 = Wf[12], *M2W2 = Wf[13];

  size_t arena = off;

  // ---- stage 1 buffers
  h16* qln1 = (h16*)alloc((size_t)98304 * 128 * 2);
  h16* qh1  = (h16*)alloc((size_t)64 * 6 * 4 * 256 * 32 * 2);
  h16* kln  = (h16*)alloc((size_t)6144 * 128 * 2);
  h16* vln  = (h16*)alloc((size_t)6144 * 128 * 2);
  h16* kh1  = (h16*)alloc((size_t)64 * 4 * 96 * 32 * 2);
  h16* v1T  = (h16*)alloc((size_t)64 * 4 * 32 * 96 * 2);
  h16* a1   = (h16*)alloc((size_t)64 * 6 * 256 * 128 * 2);
  h16* awh  = (h16*)alloc((size_t)16384 * 128 * 2);
  h16* t1   = (h16*)alloc((size_t)16384 * 128 * 2);
  h16* h1   = (h16*)alloc((size_t)16384 * 256 * 2);

  dim3 gKV(6144 / 64, 8);     // M=6144, N=128
  dim3 gT (16384 / 64, 8);    // M=16384, N=128
  dim3 gT2(16384 / 64, 16);   // M=16384, N=256
  dim3 gQ (98304 / 64, 8);    // M=98304, N=128

  k_gather_x<<<16384 * 128 / 256, 256, 0, stream>>>(x, xw);
  k_embed<<<6144, 128, 0, stream>>>(I_inv, E_inv, P(17), P(18), imge);
  k_featnorm<<<6144 * 128 / 256, 256, 0, stream>>>(feature,
      P(10), P(11), P(12), P(13), P(5), P(6), P(7), P(8), hfp, hfl);
  // key = img_embed + conv_fp(h); val = conv_fl(h)
  k_gemm<<<gKV, 128, 0, stream>>>(hfp, FPW, nullptr, imge, ktok, nullptr, 6144, 128, 128, 0, 0, 0);
  k_gemm<<<gKV, 128, 0, stream>>>(hfl, FLW, nullptr, nullptr, vtok, nullptr, 6144, 128, 128, 0, 0, 0);
  // queries: bev embed + x, LN, project+scatter to head layout
  k_qtok<<<98304, 128, 0, stream>>>(grid0, xw, E_inv, P(18), P(15), P(16), P(19), P(20), qln1);
  k_gemm<<<gQ, 128, 0, stream>>>(qln1, Wq1, P(22), nullptr, nullptr, qh1, 98304, 128, 128, 3, 1, 6);
  // keys/vals: LN, project+scatter (stage-1 contiguous windows)
  k_ln<<<6144, 128, 0, stream>>>(ktok, P(23), P(24), kln);
  k_ln<<<6144, 128, 0, stream>>>(vtok, P(27), P(28), vln);
  k_gemm<<<gKV, 128, 0, stream>>>(kln, Wk1, P(26), nullptr, nullptr, kh1, 6144, 128, 128, 4, 1, 6);
  k_gemm<<<gKV, 128, 0, stream>>>(vln, Wv1, P(30), nullptr, nullptr, v1T, 6144, 128, 128, 5, 1, 6);
  // attention per (window, camera)
  k_attn<<<dim3(64, 6), 256, 0, stream>>>(qh1, kh1, v1T, P(33), a1, 6);
  // learned camera weighting + weighted sum
  k_camweight<<<64, 128, 0, stream>>>(a1, P(34), P(35), P(36), P(37), P(38), P(39), cw);
  k_camsum<<<16384 * 128 / 256, 256, 0, stream>>>(a1, cw, awh);
  // output proj + skip(x), then MLP block
  k_gemm<<<gT, 128, 0, stream>>>(awh, Wp1, P(32), xw, out1, nullptr, 16384, 128, 128, 0, 0, 0);
  k_ln<<<16384, 128, 0, stream>>>(out1, P(61), P(62), t1);
  k_gemm<<<gT2, 128, 0, stream>>>(t1, M1W1, P(68), nullptr, nullptr, h1, 16384, 256, 128, 2, 0, 0);
  k_gemm<<<gT, 128, 0, stream>>>(h1, M1W2, P(70), out1, xs1, nullptr, 16384, 128, 256, 0, 0, 0);

  // ---- stage 2 (reuse arena; queries are camera-broadcast => camera softmax is uniform => aw == a)
  off = arena;
  h16* qln2 = (h16*)alloc((size_t)16384 * 128 * 2);
  h16* qh2  = (h16*)alloc((size_t)64 * 4 * 256 * 32 * 2);
  h16* kln2 = (h16*)alloc((size_t)6144 * 128 * 2);
  h16* vln2 = (h16*)alloc((size_t)6144 * 128 * 2);
  h16* kh2  = (h16*)alloc((size_t)64 * 4 * 96 * 32 * 2);
  h16* v2T  = (h16*)alloc((size_t)64 * 4 * 32 * 96 * 2);
  h16* a2   = (h16*)alloc((size_t)16384 * 128 * 2);
  h16* t2   = (h16*)alloc((size_t)16384 * 128 * 2);
  h16* h2   = (h16*)alloc((size_t)16384 * 256 * 2);

  k_ln<<<16384, 128, 0, stream>>>(xs1, P(40), P(41), qln2);
  k_gemm<<<gT, 128, 0, stream>>>(qln2, Wq2, P(43), nullptr, nullptr, qh2, 16384, 128, 128, 3, 2, 1);
  k_ln<<<6144, 128, 0, stream>>>(ktok, P(44), P(45), kln2);
  k_ln<<<6144, 128, 0, stream>>>(vtok, P(48), P(49), vln2);
  k_gemm<<<gKV, 128, 0, stream>>>(kln2, Wk2, P(47), nullptr, nullptr, kh2, 6144, 128, 128, 4, 2, 6);
  k_gemm<<<gKV, 128, 0, stream>>>(vln2, Wv2, P(51), nullptr, nullptr, v2T, 6144, 128, 128, 5, 2, 6);
  k_attn<<<dim3(64, 1), 256, 0, stream>>>(qh2, kh2, v2T, P(54), a2, 1);
  k_gemm<<<gT, 128, 0, stream>>>(a2, Wp2, P(53), xs1, o2a, nullptr, 16384, 128, 128, 0, 0, 0);
  k_ln<<<16384, 128, 0, stream>>>(o2a, P(63), P(64), t2);
  k_gemm<<<gT2, 128, 0, stream>>>(t2, M2W1, P(72), nullptr, nullptr, h2, 16384, 256, 128, 2, 0, 0);
  k_gemm<<<gT, 128, 0, stream>>>(h2, M2W2, P(74), o2a, o2b, nullptr, 16384, 128, 256, 0, 0, 0);
  k_final<<<16384, 128, 0, stream>>>(o2b, P(65), P(66), (float*)d_out);
}
